// DeepSeek_MLA_79078937854405
// MI455X (gfx1250) — matrix operand order repaired and checked by reading, not compiled, not run on hardware
//
#include <hip/hip_runtime.h>
#include <hip/hip_bf16.h>
#include <math.h>

// ---------------------------------------------------------------------------
// DeepSeek MLA forward for gfx1250 (CDNA5, wave32, WMMA).
// All GEMMs + attention use v_wmma_f32_16x16x32_bf16 (fp32 accumulate).
// ---------------------------------------------------------------------------

typedef __bf16 bf16_t;
typedef __attribute__((ext_vector_type(16))) __bf16 bf16x16;
typedef __attribute__((ext_vector_type(8)))  __bf16 bf16x8;
typedef __attribute__((ext_vector_type(8)))  float  f32x8;

#define Bn   2
#define Sn   2048
#define HIDn 2048
#define Hn   16
#define HDn  128
#define LATn 512
#define PDn  64
#define DQK  192           // HD + PD
#define Mrows (Bn * Sn)    // 4096

// ---------------------------------------------------------------------------
// fp32 -> bf16 elementwise convert (grid-stride)
// ---------------------------------------------------------------------------
__global__ void k_cvt_bf16(const float* __restrict__ in, bf16_t* __restrict__ out, int n) {
    int i = blockIdx.x * blockDim.x + threadIdx.x;
    int stride = gridDim.x * blockDim.x;
    for (; i < n; i += stride) out[i] = (bf16_t)in[i];
}

// ---------------------------------------------------------------------------
// Tiled bf16 GEMM: C[M,N] = A[M,K] * B[K,N], fp32 accumulate.
// Block tile 128x64, 256 threads = 8 waves in 4x2, wave tile 32x32 (2x2 WMMA).
// A staged row-major in LDS; B staged *transposed* so B-fragments are
// contiguous v16bf reads. M%128==0, N%64==0, K%32==0 for all call sites.
// ~100 VGPRs live -> allow 4 waves/EU for latency hiding.
// ---------------------------------------------------------------------------
template <bool OUT_BF16>
__global__ __launch_bounds__(256, 4)
void k_gemm(const bf16_t* __restrict__ A, const bf16_t* __restrict__ Bm,
            void* __restrict__ C, int M, int N, int K) {
    __shared__ alignas(32) bf16_t sA[128 * 32];   // [row][k]
    __shared__ alignas(32) bf16_t sBT[64 * 32];   // [n][k]

    const int tid  = threadIdx.x;
    const int lane = tid & 31;
    const int wid  = tid >> 5;
    const int lh   = lane >> 4;   // 0/1 half-wave
    const int ll   = lane & 15;

    const int tileM = blockIdx.y * 128;
    const int tileN = blockIdx.x * 64;
    const int waveM = (wid >> 1) * 32;
    const int waveN = (wid & 1) * 32;

    f32x8 acc[2][2] = {};

    for (int k0 = 0; k0 < K; k0 += 32) {
        // --- cooperative load A tile 128x32 (one v16bf per thread) ---
        {
            int row = tid >> 1;
            int cc  = (tid & 1) * 16;
            bf16x16 v = *reinterpret_cast<const bf16x16*>(
                A + (size_t)(tileM + row) * K + k0 + cc);
            *reinterpret_cast<bf16x16*>(&sA[row * 32 + cc]) = v;
        }
        // --- cooperative load B tile 32x64, transpose into sBT[n][k] ---
        {
            int row = tid >> 3;         // k within tile: 0..31
            int cc  = (tid & 7) * 8;    // n chunk
            bf16x8 v = *reinterpret_cast<const bf16x8*>(
                Bm + (size_t)(k0 + row) * N + tileN + cc);
#pragma unroll
            for (int j = 0; j < 8; ++j) sBT[(cc + j) * 32 + row] = v[j];
        }
        __syncthreads();

        // --- fragments (contiguous k = 16h + i, same order for A and B) ---
        bf16x16 af[2], bfr[2];
#pragma unroll
        for (int mi = 0; mi < 2; ++mi) {
            int m = waveM + mi * 16 + ll;
#pragma unroll
            for (int i = 0; i < 16; ++i) {
                int k = lh * 16 + i;  // contiguous k = 16h + i (matches B)
                af[mi][i] = sA[m * 32 + k];
            }
        }
#pragma unroll
        for (int ni = 0; ni < 2; ++ni) {
            int n = waveN + ni * 16 + ll;
            bfr[ni] = *reinterpret_cast<const bf16x16*>(&sBT[n * 32 + lh * 16]);
        }
#pragma unroll
        for (int mi = 0; mi < 2; ++mi)
#pragma unroll
            for (int ni = 0; ni < 2; ++ni)
                acc[mi][ni] = __builtin_amdgcn_wmma_f32_16x16x32_bf16(
                    false, af[mi], false, bfr[ni], (short)0, acc[mi][ni], false, false);
        __syncthreads();
    }

    // --- epilogue (C layout: m = (lane>>4)*8 + r, n = lane&15) ---
#pragma unroll
    for (int mi = 0; mi < 2; ++mi)
#pragma unroll
        for (int ni = 0; ni < 2; ++ni)
#pragma unroll
            for (int r = 0; r < 8; ++r) {
                int m = tileM + waveM + mi * 16 + lh * 8 + r;
                int n = tileN + waveN + ni * 16 + ll;
                float v = acc[mi][ni][r];
                if (OUT_BF16) ((bf16_t*)C)[(size_t)m * N + n] = (bf16_t)v;
                else          ((float*)C)[(size_t)m * N + n]  = v;
            }
}

// ---------------------------------------------------------------------------
// Pack + RoPE: build head-major q[BH,S,192] (pre-scaled by 1/sqrt(192)),
// k[BH,S,192], and d-major vT[BH,128,S] (so attention PV B-frags are
// contiguous). blockIdx.x = s, blockIdx.y = b*H + h, 192 threads.
// ---------------------------------------------------------------------------
__global__ __launch_bounds__(192)
void k_pack_rope(const bf16_t* __restrict__ qC, const bf16_t* __restrict__ kC,
                 const bf16_t* __restrict__ vC, const float* __restrict__ qR,
                 const float* __restrict__ kR, bf16_t* __restrict__ qP,
                 bf16_t* __restrict__ kP, bf16_t* __restrict__ vT) {
    const int s  = blockIdx.x;
    const int bh = blockIdx.y;
    const int b  = bh >> 4;    // / Hn
    const int h  = bh & 15;
    const int t  = threadIdx.x;
    const float scale = 0.07216878364870323f;  // 1/sqrt(192)

    const size_t rowBS = (size_t)b * Sn + s;
    const size_t qkRow = ((size_t)bh * Sn + s) * DQK;

    if (t < HDn) {
        float qv = (float)qC[rowBS * HIDn + (size_t)h * HDn + t];
        float kv = (float)kC[rowBS * HIDn + (size_t)h * HDn + t];
        float vv = (float)vC[rowBS * HIDn + (size_t)h * HDn + t];
        qP[qkRow + t] = (bf16_t)(qv * scale);
        kP[qkRow + t] = (bf16_t)kv;
        vT[((size_t)bh * HDn + t) * Sn + s] = (bf16_t)vv;
    } else {
        const int d  = t - HDn;      // 0..63
        const int dm = d & 31;       // freq index
        // inv_freq = 10000^(-(2*dm)/64)
        float inv_freq = __expf(-(float)(2 * dm) * (1.0f / 64.0f) * 9.210340371976184f);
        float ang = (float)s * inv_freq;
        float sn, cs;
        __sincosf(ang, &sn, &cs);
        const int   d2  = (d < 32) ? d + 32 : d - 32;
        const float sgn = (d < 32) ? -1.0f : 1.0f;

        float q1 = qR[rowBS * (Hn * PDn) + (size_t)h * PDn + d];
        float q2 = qR[rowBS * (Hn * PDn) + (size_t)h * PDn + d2];
        float k1 = kR[rowBS * PDn + d];
        float k2 = kR[rowBS * PDn + d2];
        qP[qkRow + HDn + d] = (bf16_t)((q1 * cs + sgn * q2 * sn) * scale);
        kP[qkRow + HDn + d] = (bf16_t)(k1 * cs + sgn * k2 * sn);
    }
}

// ---------------------------------------------------------------------------
// Flash attention. grid = (S/128, B*H), 256 threads = 8 waves.
// Each wave owns 16 queries; streams 32-key blocks: 12 WMMAs for Q.K^T,
// online softmax (shfl_xor row reduce across the 16-lane N groups),
// P staged through per-wave LDS into A-fragment layout, 8 WMMAs for P.V.
// Output written bf16 as [B,S,H*HD] for the final O-projection GEMM.
//
// __launch_bounds__(256, 1): ~180 VGPRs live (qf 48 + acc 64 + scores 16 +
// K frags 16 + addressing). Default occupancy heuristics spilled the resident
// Q fragments to scratch inside the key loop; wave32 permits up to 1024
// VGPRs/wave, so target 1 wave/EU min and keep everything register-resident.
// ---------------------------------------------------------------------------
__global__ __launch_bounds__(256, 1)
void k_attn(const bf16_t* __restrict__ qP, const bf16_t* __restrict__ kP,
            const bf16_t* __restrict__ vT, bf16_t* __restrict__ oB) {
    __shared__ alignas(32) bf16_t sP[8][16 * 32];  // per-wave P tile (16q x 32k)

    const int wid  = threadIdx.x >> 5;
    const int lane = threadIdx.x & 31;
    const int lh   = lane >> 4;
    const int ll   = lane & 15;

    const int bh = blockIdx.y;
    const int b  = bh >> 4;
    const int h  = bh & 15;
    const int q0 = blockIdx.x * 128 + wid * 16;

    const bf16_t* qbase = qP + ((size_t)bh * Sn + q0) * DQK;
    // per-lane base pointers (hoisted out of the key loop)
    const bf16_t* krow0 = kP + (size_t)bh * Sn * DQK + (size_t)ll * DQK + lh * 16;
    const bf16_t* vrow  = vT + (size_t)bh * HDn * Sn + (size_t)ll * Sn + lh * 16;
    bf16_t*       myP   = &sP[wid][0];

    // Q fragments: 6 chunks of K=32 along d=192, resident for whole kernel.
    bf16x16 qf[6];
#pragma unroll
    for (int kc = 0; kc < 6; ++kc) {
        bf16x8 lo = *reinterpret_cast<const bf16x8*>(
            qbase + (size_t)ll * DQK + kc * 32 + lh * 16);
        bf16x8 hi = *reinterpret_cast<const bf16x8*>(
            qbase + (size_t)ll * DQK + kc * 32 + lh * 16 + 8);
#pragma unroll
        for (int j = 0; j < 8; ++j) { qf[kc][j] = lo[j]; qf[kc][8 + j] = hi[j]; }
    }

    f32x8 acc[8] = {};
    float mrow[8], lrow[8];
#pragma unroll
    for (int r = 0; r < 8; ++r) { mrow[r] = -1e30f; lrow[r] = 0.0f; }

    for (int kb = 0; kb < Sn / 32; ++kb) {
        const size_t key0 = (size_t)kb * 32;

        // --- scores: Q(16x192) . K_tile(32x192)^T -> two 16x16 frags ---
        f32x8 s0 = {}, s1 = {};
#pragma unroll
        for (int kc = 0; kc < 6; ++kc) {
            bf16x16 b0 = *reinterpret_cast<const bf16x16*>(
                krow0 + key0 * DQK + kc * 32);
            bf16x16 b1 = *reinterpret_cast<const bf16x16*>(
                krow0 + (key0 + 16) * DQK + kc * 32);
            s0 = __builtin_amdgcn_wmma_f32_16x16x32_bf16(false, qf[kc], false, b0,
                                                         (short)0, s0, false, false);
            s1 = __builtin_amdgcn_wmma_f32_16x16x32_bf16(false, qf[kc], false, b1,
                                                         (short)0, s1, false, false);
        }

        // --- online softmax (per row r; lanes sharing a row differ only in
        //     bits [3:0], so xor masks 1/2/4/8 reduce over the N dimension) ---
#pragma unroll
        for (int r = 0; r < 8; ++r) {
            float mx = fmaxf(s0[r], s1[r]);
#pragma unroll
            for (int off = 8; off >= 1; off >>= 1)
                mx = fmaxf(mx, __shfl_xor(mx, off, 32));
            float mnew = fmaxf(mrow[r], mx);
            float corr = __expf(mrow[r] - mnew);
            float p0 = __expf(s0[r] - mnew);
            float p1 = __expf(s1[r] - mnew);
            float ps = p0 + p1;
#pragma unroll
            for (int off = 8; off >= 1; off >>= 1)
                ps += __shfl_xor(ps, off, 32);
            lrow[r] = lrow[r] * corr + ps;
            mrow[r] = mnew;
#pragma unroll
            for (int c = 0; c < 8; ++c) acc[c][r] *= corr;
            // stash P in LDS: row m = lh*8+r, cols ll and 16+ll
            myP[(lh * 8 + r) * 32 + ll]      = (bf16_t)p0;
            myP[(lh * 8 + r) * 32 + 16 + ll] = (bf16_t)p1;
        }

        // --- reload P as an A fragment (16x32); same-wave DS ops are in-order ---
        bf16x16 pa;
#pragma unroll
        for (int i = 0; i < 16; ++i) {
            int k = lh * 16 + i;
            pa[i] = myP[ll * 32 + k];
        }

        // --- P.V : 8 output chunks of 16 along dv=128; V is d-major so each
        //     B-fragment is one contiguous 32B load ---
#pragma unroll
        for (int c = 0; c < 8; ++c) {
            bf16x16 bv = *reinterpret_cast<const bf16x16*>(
                vrow + (size_t)(c * 16) * Sn + key0);
            acc[c] = __builtin_amdgcn_wmma_f32_16x16x32_bf16(false, pa, false, bv,
                                                             (short)0, acc[c], false, false);
        }
    }

    // --- normalize and store o_heads as [B,S,H*HD] bf16 ---
#pragma unroll
    for (int r = 0; r < 8; ++r) {
        float inv = 1.0f / lrow[r];
        int srow = q0 + lh * 8 + r;
        size_t orow = ((size_t)b * Sn + srow) * HIDn + (size_t)h * HDn;
#pragma unroll
        for (int c = 0; c < 8; ++c)
            oB[orow + c * 16 + ll] = (bf16_t)(acc[c][r] * inv);
    }
}

// ---------------------------------------------------------------------------
// Launcher
// ---------------------------------------------------------------------------
extern "C" void kernel_launch(void* const* d_in, const int* in_sizes, int n_in,
                              void* d_out, int out_size, void* d_ws, size_t ws_size,
                              hipStream_t stream) {
    (void)in_sizes; (void)n_in; (void)out_size; (void)ws_size;

    const float* hid   = (const float*)d_in[0];
    const float* w_dkv = (const float*)d_in[1];
    const float* w_dq  = (const float*)d_in[2];
    const float* w_uk  = (const float*)d_in[3];
    const float* w_uv  = (const float*)d_in[4];
    const float* w_uq  = (const float*)d_in[5];
    const float* w_kr  = (const float*)d_in[6];
    const float* w_qr  = (const float*)d_in[7];
    const float* w_o   = (const float*)d_in[8];
    float* out = (float*)d_out;

    char* ws = (char*)d_ws;
    size_t off = 0;
    auto wsalloc = [&](size_t bytes) -> void* {
        void* p = (void*)(ws + off);
        off += (bytes + 255) & ~(size_t)255;
        return p;
    };

    // bf16 staging buffers
    bf16_t* hbf   = (bf16_t*)wsalloc((size_t)Mrows * HIDn * 2);
    bf16_t* bdkv  = (bf16_t*)wsalloc((size_t)HIDn * LATn * 2);
    bf16_t* bdq   = (bf16_t*)wsalloc((size_t)HIDn * LATn * 2);
    bf16_t* buk   = (bf16_t*)wsalloc((size_t)LATn * HIDn * 2);
    bf16_t* buv   = (bf16_t*)wsalloc((size_t)LATn * HIDn * 2);
    bf16_t* buq   = (bf16_t*)wsalloc((size_t)LATn * HIDn * 2);
    bf16_t* bkr   = (bf16_t*)wsalloc((size_t)HIDn * PDn * 2);
    bf16_t* bqr   = (bf16_t*)wsalloc((size_t)LATn * Hn * PDn * 2);
    bf16_t* bo    = (bf16_t*)wsalloc((size_t)HIDn * HIDn * 2);
    // intermediates
    bf16_t* cKV   = (bf16_t*)wsalloc((size_t)Mrows * LATn * 2);
    bf16_t* cQ    = (bf16_t*)wsalloc((size_t)Mrows * LATn * 2);
    bf16_t* kC    = (bf16_t*)wsalloc((size_t)Mrows * HIDn * 2);
    bf16_t* vC    = (bf16_t*)wsalloc((size_t)Mrows * HIDn * 2);
    bf16_t* qC    = (bf16_t*)wsalloc((size_t)Mrows * HIDn * 2);
    float*  kRf   = (float*) wsalloc((size_t)Mrows * PDn * 4);
    float*  qRf   = (float*) wsalloc((size_t)Mrows * Hn * PDn * 4);
    bf16_t* qPack = (bf16_t*)wsalloc((size_t)Bn * Hn * Sn * DQK * 2);
    bf16_t* kPack = (bf16_t*)wsalloc((size_t)Bn * Hn * Sn * DQK * 2);
    bf16_t* vT    = (bf16_t*)wsalloc((size_t)Bn * Hn * HDn * Sn * 2);
    bf16_t* oB    = (bf16_t*)wsalloc((size_t)Mrows * HIDn * 2);

    // --- 1. fp32 -> bf16 conversions ---
    auto cvt = [&](const float* src, bf16_t* dst, size_t n) {
        int blocks = (int)((n + 255) / 256);
        if (blocks > 4096) blocks = 4096;
        k_cvt_bf16<<<blocks, 256, 0, stream>>>(src, dst, (int)n);
    };
    cvt(hid,   hbf,  (size_t)Mrows * HIDn);
    cvt(w_dkv, bdkv, (size_t)HIDn * LATn);
    cvt(w_dq,  bdq,  (size_t)HIDn * LATn);
    cvt(w_uk,  buk,  (size_t)LATn * HIDn);
    cvt(w_uv,  buv,  (size_t)LATn * HIDn);
    cvt(w_uq,  buq,  (size_t)LATn * HIDn);
    cvt(w_kr,  bkr,  (size_t)HIDn * PDn);
    cvt(w_qr,  bqr,  (size_t)LATn * Hn * PDn);
    cvt(w_o,   bo,   (size_t)HIDn * HIDn);

    // --- 2. projection GEMMs ---
    auto gemm_b = [&](const bf16_t* A, const bf16_t* B, bf16_t* C, int M, int N, int K) {
        k_gemm<true><<<dim3(N / 64, M / 128), 256, 0, stream>>>(A, B, (void*)C, M, N, K);
    };
    auto gemm_f = [&](const bf16_t* A, const bf16_t* B, float* C, int M, int N, int K) {
        k_gemm<false><<<dim3(N / 64, M / 128), 256, 0, stream>>>(A, B, (void*)C, M, N, K);
    };

    gemm_b(hbf, bdkv, cKV, Mrows, LATn, HIDn);          // c_KV
    gemm_b(hbf, bdq,  cQ,  Mrows, LATn, HIDn);          // c_Q
    gemm_f(hbf, bkr,  kRf, Mrows, PDn,  HIDn);          // k_R (pre-RoPE)
    gemm_b(cKV, buk,  kC,  Mrows, HIDn, LATn);          // k_C
    gemm_b(cKV, buv,  vC,  Mrows, HIDn, LATn);          // v_C
    gemm_b(cQ,  buq,  qC,  Mrows, HIDn, LATn);          // q_C
    gemm_f(cQ,  bqr,  qRf, Mrows, Hn * PDn, LATn);      // q_R (pre-RoPE)

    // --- 3. RoPE + head-major packing (+ V transpose, + 1/sqrt(192) on Q) ---
    k_pack_rope<<<dim3(Sn, Bn * Hn), 192, 0, stream>>>(qC, kC, vC, qRf, kRf,
                                                       qPack, kPack, vT);

    // --- 4. flash attention ---
    k_attn<<<dim3(Sn / 128, Bn * Hn), 256, 0, stream>>>(qPack, kPack, vT, oB);

    // --- 5. output projection -> fp32 d_out ---
    gemm_f(oB, bo, out, Mrows, HIDn, HIDn);
}
